// MyMultiheadAttention_37117107372359
// MI455X (gfx1250) — matrix operand order repaired and checked by reading, not compiled, not run on hardware
//
#include <hip/hip_runtime.h>
#include <hip/hip_bf16.h>
#include <math.h>

typedef __attribute__((ext_vector_type(16))) __bf16 v16bf;
typedef __attribute__((ext_vector_type(8)))  __bf16 v8bf;
typedef __attribute__((ext_vector_type(4)))  __bf16 v4bf;
typedef __attribute__((ext_vector_type(8)))  float  v8f;

#define D_MODEL 1024
#define N_HEAD  16
#define D_HEAD  64
#define BATCH   4
#define SEQ     2048
#define M_TOK   (BATCH * SEQ)   // 8192 tokens

// ---------------------------------------------------------------------------
// WMMA helpers (CDNA5 v_wmma_f32_16x16x32_bf16, wave32)
// A-frag (16x32 bf16): lane<16 -> row=lane, K=0..15 contiguous
//                      lane>=16 -> row=lane-16, K=16..31 contiguous
// B-frag (32x16 bf16): lane<16 -> col=lane, K=0..15 contiguous
//                      lane>=16 -> col=lane-16, K=16..31 contiguous
//   (B operands are always stored transposed -> one contiguous 32B read)
// C/D (16x16 f32): VGPR r: M = r + 8*(lane>=16), N = lane&15
// ---------------------------------------------------------------------------
static __device__ __forceinline__ v8f wmma_bf16(v16bf a, v16bf b, v8f c) {
  return __builtin_amdgcn_wmma_f32_16x16x32_bf16(false, a, false, b,
                                                 (short)0, c, false, false);
}

static __device__ __forceinline__ v16bf load_a16(const __bf16* p_row, int half) {
  const __bf16* p = p_row + (half ? 16 : 0);
  v8bf lo = *(const v8bf*)p;          // K +0..7   (or +16..23)
  v8bf hi = *(const v8bf*)(p + 8);    // K +8..15  (or +24..31)
  return __builtin_shufflevector(lo, hi,
      0, 1, 2, 3, 4, 5, 6, 7, 8, 9, 10, 11, 12, 13, 14, 15);
}

// ---------------------------------------------------------------------------
// Prep kernels
// ---------------------------------------------------------------------------
__global__ void cvt_f32_bf16_kernel(const float* __restrict__ src,
                                    __bf16* __restrict__ dst, int n) {
  int i = blockIdx.x * blockDim.x + threadIdx.x;
  if (i < n) dst[i] = (__bf16)src[i];
}

// src: [K][N] f32 row-major  ->  dst: [N][K] bf16 row-major
__global__ void transpose_f32_bf16_kernel(const float* __restrict__ src,
                                          __bf16* __restrict__ dst,
                                          int K, int N) {
  int i = blockIdx.x * blockDim.x + threadIdx.x;
  if (i < N * K) {
    int nn = i / K;
    int kk = i - nn * K;
    dst[i] = (__bf16)src[(size_t)kk * N + nn];
  }
}

// ---------------------------------------------------------------------------
// Register-blocked 32x64 GEMM core: one wave owns a 32x64 output tile.
// Per k-step: 2 A-frags + 4 B-frags -> 8 WMMA (each frag reused 2-4x).
// ---------------------------------------------------------------------------
struct Acc32x64 { v8f a[2][4]; };

static __device__ __forceinline__ void gemm_core_32x64(
    const __bf16* __restrict__ A,   // [M][Kdim] row-major
    const __bf16* __restrict__ BT,  // [N][Kdim] row-major (B transposed)
    int m0, int n0, int kdim, int half, int row, Acc32x64& acc) {
#pragma unroll
  for (int mi = 0; mi < 2; ++mi)
#pragma unroll
    for (int ni = 0; ni < 4; ++ni) acc.a[mi][ni] = (v8f){};

  const __bf16* ar0 = A + (size_t)(m0 + row) * kdim;
  const __bf16* ar1 = ar0 + (size_t)16 * kdim;
  const __bf16* br0 = BT + (size_t)(n0 + row) * kdim + half * 16;
  const size_t bstep = (size_t)16 * kdim;

  for (int k0 = 0; k0 < kdim; k0 += 32) {
    v16bf a0 = load_a16(ar0 + k0, half);
    v16bf a1 = load_a16(ar1 + k0, half);
#pragma unroll
    for (int ni = 0; ni < 4; ++ni) {
      v16bf bb = *(const v16bf*)(br0 + (size_t)ni * bstep + k0);
      acc.a[0][ni] = wmma_bf16(a0, bb, acc.a[0][ni]);
      acc.a[1][ni] = wmma_bf16(a1, bb, acc.a[1][ni]);
    }
  }
}

// ---------------------------------------------------------------------------
// QKV projection: [8192 x 1024] @ [1024 x 3072] + bias, scattered per head:
//   Q -> [B][H][T][Dh] bf16, K -> [B][H][T][Dh] bf16, V -> [B][H][Dh][T] bf16
// ---------------------------------------------------------------------------
__global__ void qkv_gemm_kernel(const __bf16* __restrict__ xbf,
                                const __bf16* __restrict__ wT,   // [3072][1024]
                                const float*  __restrict__ bias, // [3072]
                                __bf16* __restrict__ Qb,
                                __bf16* __restrict__ Kb,
                                __bf16* __restrict__ VTb) {
  const int lane = threadIdx.x & 31;
  const int wid  = threadIdx.x >> 5;
  const int tile = blockIdx.x * 4 + wid;
  const int TN64 = (3 * D_MODEL) / 64;        // 48 tiles along N
  const int tm = tile / TN64, tn = tile % TN64;
  const int half = lane >> 4, row = lane & 15;
  const int m0 = tm * 32, n0 = tn * 64;

  Acc32x64 acc;
  gemm_core_32x64(xbf, wT, m0, n0, D_MODEL, half, row, acc);

#pragma unroll
  for (int ni = 0; ni < 4; ++ni) {
    const int n     = n0 + ni * 16 + row;
    const int which = n >> 10;    // 0=Q 1=K 2=V (16-col group never crosses)
    const int d     = n & 1023;
    const int h     = d >> 6;
    const int dh    = d & 63;
    const float bv  = bias[n];
#pragma unroll
    for (int mi = 0; mi < 2; ++mi) {
#pragma unroll
      for (int r = 0; r < 8; ++r) {
        const int m  = m0 + mi * 16 + r + half * 8;
        const int b_ = m >> 11;          // batch
        const int t  = m & 2047;         // token
        const size_t bh = (size_t)(b_ * N_HEAD + h);
        const __bf16 val = (__bf16)(acc.a[mi][ni][r] + bv);
        if (which == 0)      Qb [(bh * SEQ + t) * D_HEAD + dh] = val;
        else if (which == 1) Kb [(bh * SEQ + t) * D_HEAD + dh] = val;
        else                 VTb[(bh * D_HEAD + dh) * SEQ + t] = val;
      }
    }
  }
}

// ---------------------------------------------------------------------------
// Fused attention per (b, h, 16-query block):
//   Phase A: S = Q·Kᵀ/8 via WMMA over causal key tiles -> LDS (bf16, 64 KB)
//   Phase B: fp32 softmax; vectorized float4 weight writes (+ zero fill of
//            the masked upper triangle); bf16 probs kept in LDS
//   Phase C: context(16x64) = P·V via WMMA, A-frags from LDS, B-frags from Vᵀ
// ---------------------------------------------------------------------------
__global__ void attn_kernel(const __bf16* __restrict__ Qb,
                            const __bf16* __restrict__ Kb,
                            const __bf16* __restrict__ VTb,
                            float*  __restrict__ wout,   // (B,H,T,T) weights
                            __bf16* __restrict__ ctx) {  // [8192][1024]
  __shared__ __align__(32) __bf16 buf[16][SEQ];          // 64 KB

  const int bid = blockIdx.x;
  const int qt  = bid & 127;
  const int h   = (bid >> 7) & 15;
  const int b   = bid >> 11;
  const int q0  = qt * 16;

  const int lane = threadIdx.x & 31;
  const int wid  = threadIdx.x >> 5;
  const int half = lane >> 4, row = lane & 15;
  const size_t bh = (size_t)(b * N_HEAD + h);
  const __bf16* Qh = Qb  + bh * (size_t)SEQ * D_HEAD;
  const __bf16* Kh = Kb  + bh * (size_t)SEQ * D_HEAD;
  const __bf16* Vt = VTb + bh * (size_t)D_HEAD * SEQ;

  // ---- Phase A: scores for key tiles 0 .. q0/16 (rest is fully masked) ----
  const int ktmax = q0 >> 4;
  const __bf16* qrow = Qh + (size_t)(q0 + row) * D_HEAD;
  v16bf aq0 = load_a16(qrow + 0, half);      // Q frags are kt-invariant
  v16bf aq1 = load_a16(qrow + 32, half);
  for (int kt = wid; kt <= ktmax; kt += 4) {
    v8f acc = {};
    const __bf16* krow = Kh + (size_t)(kt * 16 + row) * D_HEAD + half * 16;
    acc = wmma_bf16(aq0, *(const v16bf*)(krow + 0), acc);
    acc = wmma_bf16(aq1, *(const v16bf*)(krow + 32), acc);
#pragma unroll
    for (int r = 0; r < 8; ++r) {
      const int i = r + half * 8;       // local query row
      const int j = kt * 16 + row;      // key index
      float s = acc[r] * 0.125f;        // 1/sqrt(64)
      if (j > q0 + i) s = -INFINITY;    // causal mask
      buf[i][j] = (__bf16)s;
    }
  }
  __syncthreads();

  // ---- Phase B: softmax (8 threads/row, contiguous 4-elem chunks) ----
  const int rrow = threadIdx.x >> 3;
  const int sub  = threadIdx.x & 7;
  const int jend = q0 + 16;                 // first fully-masked column
  const int nk32 = (jend + 31) & ~31;       // K-extent rounded to WMMA step

  float mx = -INFINITY;
  for (int jc = sub * 4; jc < jend; jc += 32) {
#pragma unroll
    for (int e = 0; e < 4; ++e) mx = fmaxf(mx, (float)buf[rrow][jc + e]);
  }
  for (int off = 4; off >= 1; off >>= 1) mx = fmaxf(mx, __shfl_xor(mx, off, 8));
  float ssum = 0.f;
  for (int jc = sub * 4; jc < jend; jc += 32) {
#pragma unroll
    for (int e = 0; e < 4; ++e) ssum += __expf((float)buf[rrow][jc + e] - mx);
  }
  for (int off = 4; off >= 1; off >>= 1) ssum += __shfl_xor(ssum, off, 8);
  const float inv = 1.0f / ssum;

  float* wrow = wout + (bh * SEQ + (size_t)(q0 + rrow)) * SEQ;
  for (int jc = sub * 4; jc < jend; jc += 32) {
    float4 wv;
    v4bf   pv;
#pragma unroll
    for (int e = 0; e < 4; ++e) {
      const float w = __expf((float)buf[rrow][jc + e] - mx) * inv;
      (&wv.x)[e] = w;
      pv[e] = (__bf16)w;
    }
    *(float4*)(wrow + jc) = wv;             // global_store_b128
    *(v4bf*)(&buf[rrow][jc]) = pv;          // ds_store_b64
  }
  {
    const v4bf z4 = {};
    for (int jc = jend + sub * 4; jc < nk32; jc += 32)
      *(v4bf*)(&buf[rrow][jc]) = z4;        // pad LDS to WMMA K-step
    const float4 zf4 = make_float4(0.f, 0.f, 0.f, 0.f);
    for (int jc = jend + sub * 4; jc < SEQ; jc += 32)
      *(float4*)(wrow + jc) = zf4;          // causal upper triangle = 0
  }
  __syncthreads();

  // ---- Phase C: context 16x64 = P·V ; wave wid owns d0 = wid*16 ----
  {
    const int d0 = wid * 16;
    v8f acc = {};
    const __bf16* arow = &buf[row][0];                        // LDS A-frags
    const __bf16* vrow = Vt + (size_t)(d0 + row) * SEQ + half * 16;
    for (int j0 = 0; j0 < nk32; j0 += 32) {
      v16bf a  = load_a16(arow + j0, half);
      v16bf bb = *(const v16bf*)(vrow + j0);
      acc = wmma_bf16(a, bb, acc);
    }
#pragma unroll
    for (int r = 0; r < 8; ++r) {
      const int i = r + half * 8;
      ctx[((size_t)(b * SEQ) + q0 + i) * D_MODEL + h * D_HEAD + d0 + row] =
          (__bf16)acc[r];
    }
  }
}

// ---------------------------------------------------------------------------
// Output projection: out = ctx @ w_out + b_out  (fp32 result to d_out)
// ---------------------------------------------------------------------------
__global__ void out_gemm_kernel(const __bf16* __restrict__ ctx,
                                const __bf16* __restrict__ wT,   // [1024][1024]
                                const float*  __restrict__ bias,
                                float* __restrict__ out) {
  const int lane = threadIdx.x & 31;
  const int wid  = threadIdx.x >> 5;
  const int tile = blockIdx.x * 4 + wid;
  const int TN64 = D_MODEL / 64;              // 16 tiles along N
  const int tm = tile / TN64, tn = tile % TN64;
  const int half = lane >> 4, row = lane & 15;
  const int m0 = tm * 32, n0 = tn * 64;

  Acc32x64 acc;
  gemm_core_32x64(ctx, wT, m0, n0, D_MODEL, half, row, acc);

#pragma unroll
  for (int ni = 0; ni < 4; ++ni) {
    const int n    = n0 + ni * 16 + row;
    const float bv = bias[n];
#pragma unroll
    for (int mi = 0; mi < 2; ++mi) {
#pragma unroll
      for (int r = 0; r < 8; ++r) {
        out[(size_t)(m0 + mi * 16 + r + half * 8) * D_MODEL + n] =
            acc.a[mi][ni][r] + bv;
      }
    }
  }
}

// ---------------------------------------------------------------------------
extern "C" void kernel_launch(void* const* d_in, const int* in_sizes, int n_in,
                              void* d_out, int out_size, void* d_ws,
                              size_t ws_size, hipStream_t stream) {
  (void)in_sizes; (void)n_in; (void)out_size; (void)ws_size;
  const float* x     = (const float*)d_in[0];
  const float* w_qkv = (const float*)d_in[1];
  const float* b_qkv = (const float*)d_in[2];
  const float* w_out = (const float*)d_in[3];
  const float* b_out = (const float*)d_in[4];

  char* ws = (char*)d_ws;
  size_t off = 0;
  __bf16* xbf   = (__bf16*)(ws + off); off += (size_t)M_TOK * D_MODEL * 2;          // 16 MB
  __bf16* wqkvT = (__bf16*)(ws + off); off += (size_t)3 * D_MODEL * D_MODEL * 2;    //  6 MB
  __bf16* woutT = (__bf16*)(ws + off); off += (size_t)D_MODEL * D_MODEL * 2;        //  2 MB
  __bf16* Qb    = (__bf16*)(ws + off); off += (size_t)M_TOK * D_MODEL * 2;          // 16 MB
  __bf16* Kb    = (__bf16*)(ws + off); off += (size_t)M_TOK * D_MODEL * 2;          // 16 MB
  __bf16* VTb   = (__bf16*)(ws + off); off += (size_t)M_TOK * D_MODEL * 2;          // 16 MB
  __bf16* ctx   = (__bf16*)(ws + off); off += (size_t)M_TOK * D_MODEL * 2;          // 16 MB

  float* outp = (float*)d_out;                              // (B,T,D) fp32
  float* wts  = outp + (size_t)M_TOK * D_MODEL;             // (B,H,T,T) fp32

  // 1) x -> bf16
  {
    const int n = M_TOK * D_MODEL;
    cvt_f32_bf16_kernel<<<(n + 255) / 256, 256, 0, stream>>>(x, xbf, n);
  }
  // 2) weight transposes (f32 [K][N] -> bf16 [N][K])
  {
    const int n1 = 3 * D_MODEL * D_MODEL;
    transpose_f32_bf16_kernel<<<(n1 + 255) / 256, 256, 0, stream>>>(
        w_qkv, wqkvT, D_MODEL, 3 * D_MODEL);
    const int n2 = D_MODEL * D_MODEL;
    transpose_f32_bf16_kernel<<<(n2 + 255) / 256, 256, 0, stream>>>(
        w_out, woutT, D_MODEL, D_MODEL);
  }
  // 3) QKV projection + per-head scatter (V transposed); 32x64 tile per wave
  {
    const int tiles = (M_TOK / 32) * ((3 * D_MODEL) / 64); // 12288 waves
    qkv_gemm_kernel<<<tiles / 4, 128, 0, stream>>>(xbf, wqkvT, b_qkv,
                                                   Qb, Kb, VTb);
  }
  // 4) fused scores -> softmax -> weights-out -> context
  {
    const int blocks = BATCH * N_HEAD * (SEQ / 16);        // 8192
    attn_kernel<<<blocks, 128, 0, stream>>>(Qb, Kb, VTb, wts, ctx);
  }
  // 5) output projection; 32x64 tile per wave
  {
    const int tiles = (M_TOK / 32) * (D_MODEL / 64);       // 4096 waves
    out_gemm_kernel<<<tiles / 4, 128, 0, stream>>>(ctx, woutT, b_out, outp);
  }
}
